// MSDeformAttn_45767171506680
// MI455X (gfx1250) — matrix operand order repaired and checked by reading, not compiled, not run on hardware
//
#include <hip/hip_runtime.h>
#include <math.h>

// ---------------- problem constants (fixed by the reference) ----------------
#define D_MODEL   256
#define N_HEADS   8
#define N_LEVELS  4
#define N_POINTS  4
#define D_HEAD    32
#define NBATCH    2
#define NQUERY    10000
#define LEN_IN    5440   // 64*64 + 32*32 + 16*16 + 8*8

typedef __attribute__((ext_vector_type(16))) _Float16 v16h;
typedef __attribute__((ext_vector_type(8)))  _Float16 v8h;
typedef __attribute__((ext_vector_type(8)))  float    v8f;
typedef __attribute__((ext_vector_type(4)))  float    v4f;

// ---------------------------------------------------------------------------
// Elementwise f32 -> f16 conversion (vectorized x4). n must be a multiple of 4
// (all tensors here are multiples of 256 elements).
// ---------------------------------------------------------------------------
__global__ void __launch_bounds__(256)
cvt_f32_to_f16(const float* __restrict__ in, _Float16* __restrict__ out, int n4)
{
    int t = blockIdx.x * blockDim.x + threadIdx.x;
    if (t >= n4) return;
    v4f v = *(const v4f*)(in + (size_t)t * 4);
    _Float16* o = out + (size_t)t * 4;
    o[0] = (_Float16)v.x; o[1] = (_Float16)v.y;
    o[2] = (_Float16)v.z; o[3] = (_Float16)v.w;
}

// ---------------------------------------------------------------------------
// Weight convert + transpose: W[K][Nc] f32  ->  Wt[Nc][K] f16.
// Coalesced writes; tiny (<= 64K elements), L2 resident.
// ---------------------------------------------------------------------------
__global__ void __launch_bounds__(256)
cvt_transpose_w(const float* __restrict__ W, _Float16* __restrict__ Wt,
                int K, int Nc)
{
    int t = blockIdx.x * blockDim.x + threadIdx.x;
    if (t >= K * Nc) return;
    int c = t / K;          // output row    (0..Nc-1)
    int k = t % K;          // output column (0..K-1)
    Wt[(size_t)c * K + k] = (_Float16)W[(size_t)k * Nc + c];
}

// ---------------------------------------------------------------------------
// GEMM: C[M,Nc] = A[M,K=256] @ B[K,Nc] + bias[Nc],  A in f16 [M][256],
// weights pre-transposed f16 Bt[Nc][256]. One wave computes a 16x64 strip
// (4 WMMA tiles sharing one A fragment); K fully unrolled -> 32 v_wmma/wave.
// All loads are aligned 16B/32B vector loads. Tail guard is wave-uniform so
// EXEC is all-1s around every WMMA (ISA §7.12 requirement).
// ---------------------------------------------------------------------------
__global__ void __launch_bounds__(256)
gemm_wmma_f16(const _Float16* __restrict__ A,   // [M, 256]
              const _Float16* __restrict__ Bt,  // [Nc, 256]
              const float* __restrict__ bias,   // [Nc]
              float* __restrict__ C,            // [M, Nc]
              int M, int Nc)
{
    const int lane    = threadIdx.x & 31;
    const int wave    = threadIdx.x >> 5;
    const int groupsN = Nc >> 6;                       // 64-col groups
    const int nGroups = (M >> 4) * groupsN;
    const int g       = blockIdx.x * 8 + wave;
    if (g >= nGroups) return;                          // wave-uniform

    const int tm = g / groupsN;
    const int gn = g % groupsN;

    const int r  = lane & 15;
    const int kh = lane >> 4;                          // K-half select

    const _Float16* Arow = A + (size_t)(tm * 16 + r) * D_MODEL;
    const _Float16* Bc0  = Bt + (size_t)(gn * 64 +  0 + r) * D_MODEL + kh * 16;
    const _Float16* Bc1  = Bt + (size_t)(gn * 64 + 16 + r) * D_MODEL + kh * 16;
    const _Float16* Bc2  = Bt + (size_t)(gn * 64 + 32 + r) * D_MODEL + kh * 16;
    const _Float16* Bc3  = Bt + (size_t)(gn * 64 + 48 + r) * D_MODEL + kh * 16;

    v8f acc0 = {}, acc1 = {}, acc2 = {}, acc3 = {};

    #pragma unroll
    for (int k0 = 0; k0 < D_MODEL; k0 += 32) {
        // A 16x32 fragment: elems 0..15 = K k0+16*kh.. (32 contiguous bytes).
        v16h a  = *(const v16h*)(Arow + k0 + kh * 16);
        // B 32x16 fragments: elems 0..15 = K k0+16*kh.. (32 contiguous bytes).
        v16h b0 = *(const v16h*)(Bc0 + k0);
        v16h b1 = *(const v16h*)(Bc1 + k0);
        v16h b2 = *(const v16h*)(Bc2 + k0);
        v16h b3 = *(const v16h*)(Bc3 + k0);
        acc0 = __builtin_amdgcn_wmma_f32_16x16x32_f16(false, a, false, b0,
                                                      (short)0, acc0, false, false);
        acc1 = __builtin_amdgcn_wmma_f32_16x16x32_f16(false, a, false, b1,
                                                      (short)0, acc1, false, false);
        acc2 = __builtin_amdgcn_wmma_f32_16x16x32_f16(false, a, false, b2,
                                                      (short)0, acc2, false, false);
        acc3 = __builtin_amdgcn_wmma_f32_16x16x32_f16(false, a, false, b3,
                                                      (short)0, acc3, false, false);
    }

    // C/D layout: row = tm*16 + v + 8*kh, col = tile_col + (lane&15).
    const int rC = tm * 16 + kh * 8;
    #pragma unroll
    for (int j = 0; j < 4; ++j) {
        const v8f acc = (j == 0) ? acc0 : (j == 1) ? acc1 : (j == 2) ? acc2 : acc3;
        const int cC  = gn * 64 + j * 16 + r;
        const float bv = bias ? bias[cC] : 0.0f;
        #pragma unroll
        for (int v = 0; v < 8; ++v) {
            C[(size_t)(rC + v) * Nc + cC] = acc[v] + bv;
        }
    }
}

// ---------------------------------------------------------------------------
// Softmax over the last 16 (levels*points) entries, per (n,q,h).
// ---------------------------------------------------------------------------
__global__ void __launch_bounds__(256)
softmax16_kernel(float* __restrict__ attn, int total)
{
    int t = blockIdx.x * blockDim.x + threadIdx.x;
    if (t >= total) return;
    float* p = attn + (size_t)t * 16;
    float m = -INFINITY;
    #pragma unroll
    for (int i = 0; i < 16; ++i) m = fmaxf(m, p[i]);
    float s = 0.0f;
    float e[16];
    #pragma unroll
    for (int i = 0; i < 16; ++i) { e[i] = expf(p[i] - m); s += e[i]; }
    float inv = 1.0f / s;
    #pragma unroll
    for (int i = 0; i < 16; ++i) p[i] = e[i] * inv;
}

// ---------------------------------------------------------------------------
// Deformable sampling. One block per (n,q); one wave per head (h = wave id,
// d = lane): each bilinear corner is one coalesced 128 B read of a contiguous
// 32-float head slice of `value` (11 MB buffer -> L2 resident). Coordinates
// are wave-uniform (lane-redundant compute, L1 hits).
// ---------------------------------------------------------------------------
__global__ void __launch_bounds__(256)
deform_sample_kernel(const float* __restrict__ value,   // [N, LEN_IN, 256]
                     const float* __restrict__ offs,    // [N, Q, 256] (h,l,p,2)
                     const float* __restrict__ attn,    // [N, Q, 128] (h, l*p)
                     const float* __restrict__ refpts,  // [N, Q, 4, 2]
                     float* __restrict__ mid)           // [N, Q, 256]
{
    const int nq = blockIdx.x;            // n*NQUERY + q
    const int n  = nq / NQUERY;
    const int h  = threadIdx.x >> 5;
    const int d  = threadIdx.x & 31;

    const float* offp  = offs   + (size_t)nq * 256 + h * (N_LEVELS * N_POINTS * 2);
    const float* attnp = attn   + (size_t)nq * 128 + h * (N_LEVELS * N_POINTS);
    const float* refp  = refpts + (size_t)nq * (N_LEVELS * 2);

    const int lvlH[N_LEVELS] = {64, 32, 16, 8};
    const int lvlW[N_LEVELS] = {64, 32, 16, 8};
    const int lvlS[N_LEVELS] = {0, 4096, 5120, 5376};

    float acc = 0.0f;
    #pragma unroll
    for (int l = 0; l < N_LEVELS; ++l) {
        const int   Hl = lvlH[l], Wl = lvlW[l];
        const float fH = (float)Hl, fW = (float)Wl;
        const float rx = refp[l * 2 + 0];
        const float ry = refp[l * 2 + 1];
        const float* vbase =
            value + ((size_t)(n * LEN_IN + lvlS[l])) * D_MODEL + h * D_HEAD + d;

        #pragma unroll
        for (int p = 0; p < N_POINTS; ++p) {
            const float ox = offp[l * (N_POINTS * 2) + p * 2 + 0];
            const float oy = offp[l * (N_POINTS * 2) + p * 2 + 1];
            const float x  = (rx + ox / fW) * fW - 0.5f;
            const float y  = (ry + oy / fH) * fH - 0.5f;
            const float x0f = floorf(x), y0f = floorf(y);
            const int   x0 = (int)x0f,   y0 = (int)y0f;
            const float wx = x - x0f,    wy = y - y0f;

            float v00 = 0.0f, v01 = 0.0f, v10 = 0.0f, v11 = 0.0f;
            if (x0 >= 0     && x0     < Wl && y0 >= 0     && y0     < Hl)
                v00 = vbase[(size_t)(y0 * Wl + x0) * D_MODEL];
            if (x0 + 1 >= 0 && x0 + 1 < Wl && y0 >= 0     && y0     < Hl)
                v01 = vbase[(size_t)(y0 * Wl + x0 + 1) * D_MODEL];
            if (x0 >= 0     && x0     < Wl && y0 + 1 >= 0 && y0 + 1 < Hl)
                v10 = vbase[(size_t)((y0 + 1) * Wl + x0) * D_MODEL];
            if (x0 + 1 >= 0 && x0 + 1 < Wl && y0 + 1 >= 0 && y0 + 1 < Hl)
                v11 = vbase[(size_t)((y0 + 1) * Wl + x0 + 1) * D_MODEL];

            const float bil = v00 * (1.0f - wx) * (1.0f - wy)
                            + v01 * wx * (1.0f - wy)
                            + v10 * (1.0f - wx) * wy
                            + v11 * wx * wy;
            acc += attnp[l * N_POINTS + p] * bil;
        }
    }
    mid[(size_t)nq * 256 + h * D_HEAD + d] = acc;
}

// ---------------------------------------------------------------------------
extern "C" void kernel_launch(void* const* d_in, const int* in_sizes, int n_in,
                              void* d_out, int out_size, void* d_ws, size_t ws_size,
                              hipStream_t stream)
{
    (void)in_sizes; (void)n_in; (void)out_size; (void)ws_size;

    const float* query   = (const float*)d_in[0];   // [2,10000,256]
    const float* refpts  = (const float*)d_in[1];   // [2,10000,4,2]
    const float* inflat  = (const float*)d_in[2];   // [2,5440,256]
    // d_in[3], d_in[4]: spatial shapes / level starts (compile-time constants)
    const float* W_value = (const float*)d_in[5];   // [256,256]
    const float* b_value = (const float*)d_in[6];   // [256]
    const float* W_off   = (const float*)d_in[7];   // [256,256]
    const float* b_off   = (const float*)d_in[8];   // [256]
    const float* W_attn  = (const float*)d_in[9];   // [256,128]
    const float* b_attn  = (const float*)d_in[10];  // [128]
    const float* W_out   = (const float*)d_in[11];  // [256,256]
    const float* b_out   = (const float*)d_in[12];  // [256]
    float*       out     = (float*)d_out;           // [2,10000,256]

    const int Mv = NBATCH * LEN_IN;   // 10880 (16 | Mv)
    const int Mq = NBATCH * NQUERY;   // 20000 (16 | Mq)

    // ---- workspace layout -------------------------------------------------
    const size_t valueElems = (size_t)Mv * D_MODEL;   // 2,785,280
    const size_t qElems     = (size_t)Mq * D_MODEL;   // 5,120,000
    const size_t attnElems  = (size_t)Mq * 128;       // 2,560,000

    float* ws_value = (float*)d_ws;                   // f32 value
    float* ws_off   = ws_value + valueElems;          // f32 offsets
    float* ws_attn  = ws_off   + qElems;              // f32 attn
    float* ws_mid   = ws_attn  + attnElems;           // f32 sampled mid
    _Float16* h_inflat = (_Float16*)(ws_mid + qElems);     // f16 input_flatten
    _Float16* h_query  = h_inflat + valueElems;            // f16 query
    _Float16* h_mid    = h_query  + qElems;                // f16 mid
    _Float16* h_Wv     = h_mid    + qElems;                // f16 W_value^T [256][256]
    _Float16* h_Wo     = h_Wv     + 256 * 256;             // f16 W_off^T   [256][256]
    _Float16* h_Wa     = h_Wo     + 256 * 256;             // f16 W_attn^T  [128][256]
    _Float16* h_Wu     = h_Wa     + 128 * 256;             // f16 W_out^T   [256][256]

    // ---- 0) one-time f16 conversions (L2-resident elementwise passes) -----
    cvt_f32_to_f16<<<(int)((valueElems / 4 + 255) / 256), 256, 0, stream>>>(
        inflat, h_inflat, (int)(valueElems / 4));
    cvt_f32_to_f16<<<(int)((qElems / 4 + 255) / 256), 256, 0, stream>>>(
        query, h_query, (int)(qElems / 4));
    cvt_transpose_w<<<(256 * 256 + 255) / 256, 256, 0, stream>>>(W_value, h_Wv, 256, 256);
    cvt_transpose_w<<<(256 * 256 + 255) / 256, 256, 0, stream>>>(W_off,   h_Wo, 256, 256);
    cvt_transpose_w<<<(256 * 128 + 255) / 256, 256, 0, stream>>>(W_attn,  h_Wa, 256, 128);
    cvt_transpose_w<<<(256 * 256 + 255) / 256, 256, 0, stream>>>(W_out,   h_Wu, 256, 256);

    // ---- 1) value = input_flatten @ W_value + b_value ---------------------
    {
        int groups = (Mv / 16) * (256 / 64);      // 2720 -> 340 blocks
        gemm_wmma_f16<<<(groups + 7) / 8, 256, 0, stream>>>(
            h_inflat, h_Wv, b_value, ws_value, Mv, 256);
    }
    // ---- 2) off = query @ W_off + b_off -----------------------------------
    {
        int groups = (Mq / 16) * (256 / 64);      // 5000 -> 625 blocks
        gemm_wmma_f16<<<(groups + 7) / 8, 256, 0, stream>>>(
            h_query, h_Wo, b_off, ws_off, Mq, 256);
    }
    // ---- 3) attn logits = query @ W_attn + b_attn -------------------------
    {
        int groups = (Mq / 16) * (128 / 64);      // 2500 -> 313 blocks
        gemm_wmma_f16<<<(groups + 7) / 8, 256, 0, stream>>>(
            h_query, h_Wa, b_attn, ws_attn, Mq, 128);
    }
    // ---- 4) softmax over 16 (level,point) logits per (n,q,h) --------------
    {
        int total = Mq * N_HEADS;                 // 160000
        softmax16_kernel<<<(total + 255) / 256, 256, 0, stream>>>(ws_attn, total);
    }
    // ---- 5) deformable bilinear sampling -> mid ---------------------------
    deform_sample_kernel<<<NBATCH * NQUERY, 256, 0, stream>>>(
        ws_value, ws_off, ws_attn, refpts, ws_mid);

    // ---- 6) out = mid @ W_out + b_out -------------------------------------
    cvt_f32_to_f16<<<(int)((qElems / 4 + 255) / 256), 256, 0, stream>>>(
        ws_mid, h_mid, (int)(qElems / 4));
    {
        int groups = (Mq / 16) * (256 / 64);      // 5000 -> 625 blocks
        gemm_wmma_f16<<<(groups + 7) / 8, 256, 0, stream>>>(
            h_mid, h_Wu, b_out, out, Mq, 256);
    }
}
